// BiLSTM_CRF_25933012533512
// MI455X (gfx1250) — matrix operand order repaired and checked by reading, not compiled, not run on hardware
//
#include <hip/hip_runtime.h>
#include <stdint.h>

// ---------------------------------------------------------------------------
// BiLSTM-CRF for MI455X (gfx1250, wave32).
//   K_cvt   : f32 -> bf16 weight conversion
//   K_gather: embedding gather + f32->bf16
//   K_gemm  : bf16 WMMA GEMM, each wave computes a 16x64 strip (4 accumulators,
//             A fragment reused across 4 v_wmma_f32_16x16x32_bf16)
//   K_lstm  : sequential recurrence, one block per direction; Z[t] rows are
//             double-buffered into LDS by the Tensor Data Mover
//             (tensor_load_to_lds + s_wait_tensorcnt), W_hh streamed as bf16
//   K_vit   : Viterbi forward (trans in LDS) + single-thread backtrack
// ---------------------------------------------------------------------------

#define LSEQ 4096
#define EDIM 512
#define HDIM 512
#define GDIM 2048   // 4*HDIM
#define TTAG 64

typedef __attribute__((ext_vector_type(16))) __bf16 v16bf;
typedef __attribute__((ext_vector_type(8)))  float  v8f;
typedef __attribute__((ext_vector_type(4)))  unsigned int v4u;
typedef __attribute__((ext_vector_type(8)))  int    v8i;
typedef __attribute__((ext_vector_type(4)))  int    v4i;

#if defined(__has_builtin)
#  if __has_builtin(__builtin_amdgcn_tensor_load_to_lds) && \
      __has_builtin(__builtin_amdgcn_s_wait_tensorcnt)
#    define USE_TDM 1
#  endif
#endif
#ifndef USE_TDM
#  define USE_TDM 0
#endif

__device__ __forceinline__ unsigned short f2bf(float f) {
    union { float f; unsigned u; } v; v.f = f;
    unsigned r = v.u + 0x7fffu + ((v.u >> 16) & 1u);   // round-to-nearest-even
    return (unsigned short)(r >> 16);
}
__device__ __forceinline__ float bf2f(unsigned short s) {
    union { unsigned u; float f; } v; v.u = ((unsigned)s) << 16; return v.f;
}
__device__ __forceinline__ float sigf(float x) { return 1.0f / (1.0f + __expf(-x)); }

#if USE_TDM
// TDM: copy `nelem` f32 from global `gsrc` into LDS offset `lds_off`.
// D# per CDNA5 ISA:
//   group0: count=1 | lds_addr | global_addr | type=2
//   group1: data_size=2 (4B), tensor_dim0=tile_dim0=nelem, tensor_dim1=tile_dim1=1,
//           tensor_dim0_stride=nelem, workgroup_mask=0
// clang-23 toolchain builtin arity: (v4u, v8i, v4i, v4i, v8i, i32 cpol)
__device__ __forceinline__ void tdm_load_f32(unsigned lds_off, const float* gsrc, int nelem) {
    unsigned long long ga = (unsigned long long)(uintptr_t)gsrc;
    v4u g0;
    g0[0] = 1u;                                              // count=1, no gather
    g0[1] = lds_off;                                         // lds_addr
    g0[2] = (unsigned)(ga & 0xffffffffu);                    // global_addr[31:0]
    g0[3] = (unsigned)((ga >> 32) & 0x01ffffffu) | 0x80000000u; // addr[56:32] | type=2
    v8i g1;
    g1[0] = (int)(2u << 16);                                 // data_size = 4 bytes
    g1[1] = (int)(((unsigned)nelem & 0xffffu) << 16);        // tensor_dim0[15:0] @ [63:48]
    g1[2] = (int)((((unsigned)nelem >> 16) & 0xffffu) | (1u << 16)); // dim0 hi | tensor_dim1=1
    g1[3] = (int)(((unsigned)nelem & 0xffffu) << 16);        // tile_dim0 @ [127:112]
    g1[4] = 1;                                               // tile_dim1 = 1
    g1[5] = nelem;                                           // tensor_dim0_stride[31:0]
    g1[6] = 0;
    g1[7] = 0;
    v4i z4 = {0, 0, 0, 0};
    v8i z8 = {0, 0, 0, 0, 0, 0, 0, 0};
    __builtin_amdgcn_tensor_load_to_lds(g0, g1, z4, z4, z8, 0);
}
#endif

// -------------------------- f32 -> bf16 convert ----------------------------
__global__ void K_cvt(const float* __restrict__ src, unsigned short* __restrict__ dst, int n) {
    int i = blockIdx.x * blockDim.x + threadIdx.x;
    if (i < n) dst[i] = f2bf(src[i]);
}

// ---------------------- embedding gather + convert -------------------------
__global__ void K_gather(const int* __restrict__ sent, const float* __restrict__ emb,
                         unsigned short* __restrict__ xbf) {
    int i = blockIdx.x * blockDim.x + threadIdx.x;          // over LSEQ*EDIM
    if (i >= LSEQ * EDIM) return;
    int t = i >> 9, e = i & (EDIM - 1);
    xbf[i] = f2bf(emb[(long)sent[t] * EDIM + e]);
}

// --------------------------- bf16 WMMA GEMM --------------------------------
// C[M,N] = A[M,K] (row-major bf16) * B[N,K]^T (row-major bf16) + bias[N]
// One 16(M) x 64(N) strip per wave: A fragment reused across 4 WMMAs per
// K-chunk of 32. Requires N % 64 == 0, K % 32 == 0.
__global__ void K_gemm(const unsigned short* __restrict__ A,
                       const unsigned short* __restrict__ B,
                       const float* __restrict__ bias,
                       float* __restrict__ C,
                       int M, int N, int K) {
    int lane   = threadIdx.x & 31;
    int wave   = threadIdx.x >> 5;
    int tile   = blockIdx.x * (blockDim.x >> 5) + wave;
    int ngrp   = N >> 6;
    int total  = (M >> 4) * ngrp;
    if (tile >= total) return;
    int mtile = tile / ngrp, ntile = tile % ngrp;
    int row0 = mtile << 4, col0 = ntile << 6;
    int half = lane >> 4, l16 = lane & 15;

    const unsigned short* arow = A + (long)(row0 + l16) * K;
    const unsigned short* br0  = B + (long)(col0 + l16) * K;        // strip cols 0..15
    const unsigned short* br1  = br0 + (long)16 * K;                // 16..31
    const unsigned short* br2  = br0 + (long)32 * K;                // 32..47
    const unsigned short* br3  = br0 + (long)48 * K;                // 48..63

    v8f c0 = {}, c1 = {}, c2 = {}, c3 = {};
    for (int kb = 0; kb < K; kb += 32) {
        v16bf a, b0, b1, b2, b3;
        {   // A: contiguous 16 values K = kb + half*16 .. +15, row = row0 + l16
            uint4* av = reinterpret_cast<uint4*>(&a);
            av[0] = *reinterpret_cast<const uint4*>(arow + kb + half * 16);
            av[1] = *reinterpret_cast<const uint4*>(arow + kb + half * 16 + 8);
        }
        {   // B: contiguous 16 values K = kb + half*16 .. +15, col = group base + l16
            const uint4* p;
            uint4* bv;
            p = reinterpret_cast<const uint4*>(br0 + kb + half * 16);
            bv = reinterpret_cast<uint4*>(&b0); bv[0] = p[0]; bv[1] = p[1];
            p = reinterpret_cast<const uint4*>(br1 + kb + half * 16);
            bv = reinterpret_cast<uint4*>(&b1); bv[0] = p[0]; bv[1] = p[1];
            p = reinterpret_cast<const uint4*>(br2 + kb + half * 16);
            bv = reinterpret_cast<uint4*>(&b2); bv[0] = p[0]; bv[1] = p[1];
            p = reinterpret_cast<const uint4*>(br3 + kb + half * 16);
            bv = reinterpret_cast<uint4*>(&b3); bv[0] = p[0]; bv[1] = p[1];
        }
        c0 = __builtin_amdgcn_wmma_f32_16x16x32_bf16(false, a, false, b0, (short)0, c0, false, false);
        c1 = __builtin_amdgcn_wmma_f32_16x16x32_bf16(false, a, false, b1, (short)0, c1, false, false);
        c2 = __builtin_amdgcn_wmma_f32_16x16x32_bf16(false, a, false, b2, (short)0, c2, false, false);
        c3 = __builtin_amdgcn_wmma_f32_16x16x32_bf16(false, a, false, b3, (short)0, c3, false, false);
    }
    // C/D layout: VGPR r = D[M = half*8 + r][N = lane&15]
    float* crow = C + (long)(row0 + half * 8) * N + col0 + l16;
    float bs0 = bias[col0 + l16];
    float bs1 = bias[col0 + 16 + l16];
    float bs2 = bias[col0 + 32 + l16];
    float bs3 = bias[col0 + 48 + l16];
    #pragma unroll
    for (int r = 0; r < 8; ++r) {
        crow[(long)r * N +  0] = c0[r] + bs0;
        crow[(long)r * N + 16] = c1[r] + bs1;
        crow[(long)r * N + 32] = c2[r] + bs2;
        crow[(long)r * N + 48] = c3[r] + bs3;
    }
}

// ------------------------------ LSTM recurrence ----------------------------
// blockIdx.x = direction (0 fwd, 1 bwd). 1024 threads. Thread tid computes
// gate rows {tid, tid+1024}; threads <512 then combine (i,f,g,o) -> (c,h).
// Z[t] rows (8KB f32) are double-buffered into LDS by TDM one step ahead.
__global__ void K_lstm(const float* __restrict__ Zf, const float* __restrict__ Zb,
                       const unsigned short* __restrict__ Wf,
                       const unsigned short* __restrict__ Wb,
                       unsigned short* __restrict__ hout /* [LSEQ][2*HDIM] bf16 */) {
    __shared__ float hsh[HDIM];
    __shared__ float zbuf[GDIM];
#if USE_TDM
    __shared__ float zin[2][GDIM];
#endif
    const int tid = threadIdx.x;
    const int dir = blockIdx.x;
    const float* Z = dir ? Zb : Zf;
    const unsigned short* W = dir ? Wb : Wf;
    const int hoff = dir ? HDIM : 0;

    if (tid < HDIM) hsh[tid] = 0.0f;
    float cst = 0.0f;                       // c state (threads < 512)

    const unsigned short* w0 = W + (long)tid * EDIM;          // gate row tid
    const unsigned short* w1 = W + (long)(tid + 1024) * EDIM; // gate row tid+1024

#if USE_TDM
    int cur = 0;
    {   // preload Z[t0] (wave 0 issues the DMA; TENSORcnt is per-wave)
        int t0 = dir ? (LSEQ - 1) : 0;
        if (tid < 32) {
            tdm_load_f32((unsigned)(uintptr_t)&zin[0][0], Z + (long)t0 * GDIM, GDIM);
            __builtin_amdgcn_s_wait_tensorcnt(0);
        }
    }
#endif
    __syncthreads();

    for (int s = 0; s < LSEQ; ++s) {
        int t = dir ? (LSEQ - 1 - s) : s;
#if USE_TDM
        // kick off DMA for the next step's Z row while we compute this one
        if (s + 1 < LSEQ && tid < 32) {
            int tn = dir ? (LSEQ - 2 - s) : (s + 1);
            tdm_load_f32((unsigned)(uintptr_t)&zin[cur ^ 1][0], Z + (long)tn * GDIM, GDIM);
        }
        float a0 = zin[cur][tid], a1 = zin[cur][tid + 1024];
#else
        const float* zrow = Z + (long)t * GDIM;
        float a0 = zrow[tid], a1 = zrow[tid + 1024];
#endif
        #pragma unroll 4
        for (int kb = 0; kb < HDIM; kb += 8) {
            uint4 ua = *reinterpret_cast<const uint4*>(w0 + kb);
            uint4 ub = *reinterpret_cast<const uint4*>(w1 + kb);
            const unsigned short* pa = reinterpret_cast<const unsigned short*>(&ua);
            const unsigned short* pb = reinterpret_cast<const unsigned short*>(&ub);
            #pragma unroll
            for (int j = 0; j < 8; ++j) {
                float hv = hsh[kb + j];
                a0 = __builtin_fmaf(bf2f(pa[j]), hv, a0);
                a1 = __builtin_fmaf(bf2f(pb[j]), hv, a1);
            }
        }
        zbuf[tid] = a0; zbuf[tid + 1024] = a1;
        __syncthreads();
        if (tid < HDIM) {
            float ig = sigf(zbuf[tid]);
            float fg = sigf(zbuf[tid + 512]);
            float gg = tanhf(zbuf[tid + 1024]);
            float og = sigf(zbuf[tid + 1536]);
            cst = fg * cst + ig * gg;
            float hv = og * tanhf(cst);
            hsh[tid] = hv;
            hout[(long)t * (2 * HDIM) + hoff + tid] = f2bf(hv);
        }
#if USE_TDM
        if (tid < 32) __builtin_amdgcn_s_wait_tensorcnt(0);  // next Z row landed
#endif
        __syncthreads();
#if USE_TDM
        cur ^= 1;
#endif
    }
}

// ------------------------------- Viterbi -----------------------------------
__global__ void K_vit(const float* __restrict__ feats,   // [LSEQ][TTAG]
                      const float* __restrict__ trans,   // [TTAG][TTAG]
                      const float* __restrict__ start_t,
                      const float* __restrict__ end_t,
                      const int* __restrict__ pb, const int* __restrict__ pe,
                      int* __restrict__ bp,              // [LSEQ][TTAG]
                      float* __restrict__ out) {
    __shared__ float tr[TTAG * TTAG];
    __shared__ float sc[TTAG], nsc[TTAG];
    const int j = threadIdx.x;
    const int b = pb[0];
    const int n = pe[0] - b;
    for (int i = j; i < TTAG * TTAG; i += TTAG) tr[i] = trans[i];
    sc[j] = start_t[j] + feats[(long)b * TTAG + j];
    __syncthreads();

    for (int t = 1; t < n; ++t) {
        float best = -3.4e38f; int bi = 0;
        #pragma unroll 8
        for (int i = 0; i < TTAG; ++i) {
            float v = sc[i] + tr[i * TTAG + j];
            if (v > best) { best = v; bi = i; }
        }
        nsc[j] = best + feats[(long)(b + t) * TTAG + j];
        bp[(long)t * TTAG + j] = bi;
        __syncthreads();
        sc[j] = nsc[j];
        __syncthreads();
    }
    sc[j] += end_t[j];
    __threadfence();
    __syncthreads();
    if (j == 0) {
        int best = 0; float bv = sc[0];
        for (int i = 1; i < TTAG; ++i) if (sc[i] > bv) { bv = sc[i]; best = i; }
        int tag = best;
        out[n - 1] = (float)tag;
        for (int k = n - 2; k >= 0; --k) {
            tag = bp[(long)(k + 1) * TTAG + tag];
            out[k] = (float)tag;
        }
    }
}

// ------------------------------- launcher ----------------------------------
extern "C" void kernel_launch(void* const* d_in, const int* in_sizes, int n_in,
                              void* d_out, int out_size, void* d_ws, size_t ws_size,
                              hipStream_t stream) {
    const int*   sent   = (const int*)  d_in[0];
    const int*   pb     = (const int*)  d_in[1];
    const int*   pe     = (const int*)  d_in[2];
    const float* emb    = (const float*)d_in[3];
    const float* Wih_f  = (const float*)d_in[4];
    const float* Whh_f  = (const float*)d_in[5];
    const float* b_f    = (const float*)d_in[6];
    const float* Wih_b  = (const float*)d_in[7];
    const float* Whh_b  = (const float*)d_in[8];
    const float* b_b    = (const float*)d_in[9];
    const float* fc_w   = (const float*)d_in[10];
    const float* fc_b   = (const float*)d_in[11];
    const float* startt = (const float*)d_in[12];
    const float* endt   = (const float*)d_in[13];
    const float* trans  = (const float*)d_in[14];

    char* ws = (char*)d_ws;
    size_t off = 0;
    auto alloc = [&](size_t bytes) { char* p = ws + off; off += (bytes + 255) & ~(size_t)255; return p; };
    unsigned short* wihf_bf = (unsigned short*)alloc((size_t)GDIM * EDIM * 2);
    unsigned short* wihb_bf = (unsigned short*)alloc((size_t)GDIM * EDIM * 2);
    unsigned short* whhf_bf = (unsigned short*)alloc((size_t)GDIM * HDIM * 2);
    unsigned short* whhb_bf = (unsigned short*)alloc((size_t)GDIM * HDIM * 2);
    unsigned short* fcw_bf  = (unsigned short*)alloc((size_t)TTAG * 2 * HDIM * 2);
    unsigned short* x_bf    = (unsigned short*)alloc((size_t)LSEQ * EDIM * 2);
    float*          Zf      = (float*)         alloc((size_t)LSEQ * GDIM * 4);
    float*          Zb      = (float*)         alloc((size_t)LSEQ * GDIM * 4);
    unsigned short* h_bf    = (unsigned short*)alloc((size_t)LSEQ * 2 * HDIM * 2);
    float*          feats   = (float*)         alloc((size_t)LSEQ * TTAG * 4);
    int*            bp      = (int*)           alloc((size_t)LSEQ * TTAG * 4);
    (void)ws_size; (void)n_in; (void)in_sizes; (void)out_size;

    // 1) convert weights to bf16
    const int CT = 256;
    int nW = GDIM * EDIM;
    K_cvt<<<(nW + CT - 1) / CT, CT, 0, stream>>>(Wih_f, wihf_bf, nW);
    K_cvt<<<(nW + CT - 1) / CT, CT, 0, stream>>>(Wih_b, wihb_bf, nW);
    K_cvt<<<(nW + CT - 1) / CT, CT, 0, stream>>>(Whh_f, whhf_bf, nW);
    K_cvt<<<(nW + CT - 1) / CT, CT, 0, stream>>>(Whh_b, whhb_bf, nW);
    int nF = TTAG * 2 * HDIM;
    K_cvt<<<(nF + CT - 1) / CT, CT, 0, stream>>>(fc_w, fcw_bf, nF);

    // 2) embedding gather -> bf16
    int nX = LSEQ * EDIM;
    K_gather<<<(nX + CT - 1) / CT, CT, 0, stream>>>(sent, emb, x_bf);

    // 3) input projections via WMMA: Z = x @ W_ih^T + b   (M=4096, N=2048, K=512)
    {
        int tiles  = (LSEQ / 16) * (GDIM / 64);      // 16x64 strip per wave
        int blocks = (tiles + 7) / 8;                // 8 waves / block
        K_gemm<<<blocks, 256, 0, stream>>>(x_bf, wihf_bf, b_f, Zf, LSEQ, GDIM, EDIM);
        K_gemm<<<blocks, 256, 0, stream>>>(x_bf, wihb_bf, b_b, Zb, LSEQ, GDIM, EDIM);
    }

    // 4) recurrence: block 0 forward, block 1 backward (independent)
    K_lstm<<<2, 1024, 0, stream>>>(Zf, Zb, whhf_bf, whhb_bf, h_bf);

    // 5) fc via WMMA: feats = h @ fc_w^T + fc_b   (M=4096, N=64, K=1024)
    {
        int tiles  = (LSEQ / 16) * (TTAG / 64);      // 256 strips
        int blocks = (tiles + 7) / 8;
        K_gemm<<<blocks, 256, 0, stream>>>(h_bf, fcw_bf, fc_b, feats, LSEQ, TTAG, 2 * HDIM);
    }

    // 6) Viterbi + backtrack
    K_vit<<<1, TTAG, 0, stream>>>(feats, trans, startt, endt, pb, pe, bp, (float*)d_out);
}
